// CrossAttentionLayer_62560493633971
// MI455X (gfx1250) — matrix operand order repaired and checked by reading, not compiled, not run on hardware
//
#include <hip/hip_runtime.h>
#include <hip/hip_bf16.h>

typedef __bf16 bf16;
typedef __attribute__((ext_vector_type(16))) __bf16 v16bf;
typedef __attribute__((ext_vector_type(8)))  __bf16 v8bf;
typedef __attribute__((ext_vector_type(4)))  __bf16 v4bf;
typedef __attribute__((ext_vector_type(8)))  float  v8f;
typedef __attribute__((ext_vector_type(4)))  int    i32x4;

#define BATCH 8
#define CH    512
#define NPIX  4096   // 64*64
#define HID   64

// ---- CDNA5 async load-to-LDS path (guarded; sync fallback keeps compile clean) ----
#if defined(__has_builtin)
#  if __has_builtin(__builtin_amdgcn_global_load_async_to_lds_b128)
#    define HAVE_ASYNC_LDS 1
#  else
#    define HAVE_ASYNC_LDS 0
#  endif
#else
#  define HAVE_ASYNC_LDS 0
#endif

#if HAVE_ASYNC_LDS
#  define AS1 __attribute__((address_space(1)))
#  define AS3 __attribute__((address_space(3)))
#  if __has_builtin(__builtin_amdgcn_s_wait_asynccnt)
#    define WAIT_ASYNC(n) __builtin_amdgcn_s_wait_asynccnt(n)
#  else
#    define WAIT_ASYNC(n) asm volatile("s_wait_asynccnt %0" :: "i"(n) : "memory")
#  endif
#else
#  define WAIT_ASYNC(n) do { } while (0)
#endif

__device__ __forceinline__ void async_copy16(const bf16* g, bf16* l) {
#if HAVE_ASYNC_LDS
  __builtin_amdgcn_global_load_async_to_lds_b128(
      (AS1 i32x4*)g, (AS3 i32x4*)l, 0, 0);
#else
  *(v8bf*)l = *(const v8bf*)g;   // sync fallback: global_load_b128 + ds_store_b128
#endif
}

__device__ __forceinline__ v8f vzero8() {
  v8f z = {0.f,0.f,0.f,0.f,0.f,0.f,0.f,0.f};
  return z;
}

// concatenate two 8-element bf16 vectors loaded 32B apart into a 16-elem fragment
__device__ __forceinline__ v16bf frag_cat(const bf16* p0, const bf16* p1) {
  v8bf a = *(const v8bf*)p0;
  v8bf b = *(const v8bf*)p1;
  return __builtin_shufflevector(a, b, 0,1,2,3,4,5,6,7,8,9,10,11,12,13,14,15);
}

__device__ __forceinline__ v8f wmma_bf16(v16bf a, v16bf b, v8f c) {
  return __builtin_amdgcn_wmma_f32_16x16x32_bf16(false, a, false, b, (short)0, c, false, false);
}

__device__ __forceinline__ float red16_max(float v) {
  #pragma unroll
  for (int m = 1; m < 16; m <<= 1) v = fmaxf(v, __shfl_xor(v, m, 32));
  return v;
}
__device__ __forceinline__ float red16_sum(float v) {
  #pragma unroll
  for (int m = 1; m < 16; m <<= 1) v += __shfl_xor(v, m, 32);
  return v;
}

// ---------------- cast kernels ----------------

__global__ void k_cast_v(const float* __restrict__ kf, bf16* __restrict__ Vb) {
  long i = (long)blockIdx.x * blockDim.x + threadIdx.x;     // quad index
  const long total = (long)BATCH * CH * NPIX / 4;
  if (i >= total) return;
  float4 x = ((const float4*)kf)[i];
  v4bf o; o[0] = (bf16)x.x; o[1] = (bf16)x.y; o[2] = (bf16)x.z; o[3] = (bf16)x.w;
  ((v4bf*)Vb)[i] = o;
}

__global__ void k_cast_w(const float* __restrict__ Wq, const float* __restrict__ Wk,
                         bf16* __restrict__ Wqb, bf16* __restrict__ Wkb) {
  int i = blockIdx.x * blockDim.x + threadIdx.x;            // quad index over 2*8192
  const int per = HID * CH / 4;
  if (i >= 2 * per) return;
  const float4* src = (i < per) ? (const float4*)Wq : (const float4*)Wk;
  v4bf* dst = (i < per) ? (v4bf*)Wqb : (v4bf*)Wkb;
  int j = (i < per) ? i : i - per;
  float4 x = src[j];
  v4bf o; o[0] = (bf16)x.x; o[1] = (bf16)x.y; o[2] = (bf16)x.z; o[3] = (bf16)x.w;
  dst[j] = o;
}

// ---------------- projection: Q = Wq*X + bq, K = Wk*X + bk ----------------
__global__ void __launch_bounds__(256) k_proj(
    const float* __restrict__ Xq, const float* __restrict__ Xk,
    const bf16* __restrict__ Wqb, const bf16* __restrict__ Wkb,
    const float* __restrict__ biasq, const float* __restrict__ biask,
    bf16* __restrict__ Qb, bf16* __restrict__ Kb)
{
  const int bidx  = blockIdx.y;
  const int nbase = blockIdx.x * 128;
  const int tid   = threadIdx.x;
  const int wave  = tid >> 5;
  const int lane  = tid & 31;
  const int half  = lane >> 4;
  const int l15   = lane & 15;

  __shared__ bf16 ldsq[128 * 40];   // [n_local][c_local], padded stride 40
  __shared__ bf16 ldsk[128 * 40];

  v8f accQ[4], accK[4];
  #pragma unroll
  for (int t = 0; t < 4; ++t) { accQ[t] = vzero8(); accK[t] = vzero8(); }

  for (int c0 = 0; c0 < CH; c0 += 32) {
    #pragma unroll
    for (int i = 0; i < 4; ++i) {
      int p   = tid + i * 256;          // 0..1023 quad slots
      int c_l = p >> 5;                 // 0..31
      int n_l = (p & 31) << 2;          // 0..124
      long g  = ((long)bidx * CH + (c0 + c_l)) * NPIX + nbase + n_l;
      float4 xq = *(const float4*)(Xq + g);
      float4 xk = *(const float4*)(Xk + g);
      ldsq[(n_l+0)*40 + c_l] = (bf16)xq.x;
      ldsq[(n_l+1)*40 + c_l] = (bf16)xq.y;
      ldsq[(n_l+2)*40 + c_l] = (bf16)xq.z;
      ldsq[(n_l+3)*40 + c_l] = (bf16)xq.w;
      ldsk[(n_l+0)*40 + c_l] = (bf16)xk.x;
      ldsk[(n_l+1)*40 + c_l] = (bf16)xk.y;
      ldsk[(n_l+2)*40 + c_l] = (bf16)xk.z;
      ldsk[(n_l+3)*40 + c_l] = (bf16)xk.w;
    }
    __syncthreads();
    {
      int nloc = wave * 16 + l15;
      const bf16* pq = &ldsq[nloc * 40 + half * 8];
      const bf16* pk = &ldsk[nloc * 40 + half * 8];
      v16bf Bq = frag_cat(pq, pq + 16);
      v16bf Bk = frag_cat(pk, pk + 16);
      #pragma unroll
      for (int t = 0; t < 4; ++t) {
        int d = t * 16 + l15;
        const bf16* wq = Wqb + (long)d * CH + c0 + half * 8;
        const bf16* wk = Wkb + (long)d * CH + c0 + half * 8;
        v16bf Aq = frag_cat(wq, wq + 16);
        v16bf Ak = frag_cat(wk, wk + 16);
        accQ[t] = wmma_bf16(Aq, Bq, accQ[t]);
        accK[t] = wmma_bf16(Ak, Bk, accK[t]);
      }
    }
    __syncthreads();
  }

  int n = nbase + wave * 16 + l15;
  #pragma unroll
  for (int t = 0; t < 4; ++t) {
    int d0 = t * 16 + half * 8;
    float4 bq0 = *(const float4*)(biasq + d0);
    float4 bq1 = *(const float4*)(biasq + d0 + 4);
    float4 bk0 = *(const float4*)(biask + d0);
    float4 bk1 = *(const float4*)(biask + d0 + 4);
    v8bf oq, ok;
    oq[0]=(bf16)(accQ[t][0]+bq0.x); oq[1]=(bf16)(accQ[t][1]+bq0.y);
    oq[2]=(bf16)(accQ[t][2]+bq0.z); oq[3]=(bf16)(accQ[t][3]+bq0.w);
    oq[4]=(bf16)(accQ[t][4]+bq1.x); oq[5]=(bf16)(accQ[t][5]+bq1.y);
    oq[6]=(bf16)(accQ[t][6]+bq1.z); oq[7]=(bf16)(accQ[t][7]+bq1.w);
    ok[0]=(bf16)(accK[t][0]+bk0.x); ok[1]=(bf16)(accK[t][1]+bk0.y);
    ok[2]=(bf16)(accK[t][2]+bk0.z); ok[3]=(bf16)(accK[t][3]+bk0.w);
    ok[4]=(bf16)(accK[t][4]+bk1.x); ok[5]=(bf16)(accK[t][5]+bk1.y);
    ok[6]=(bf16)(accK[t][6]+bk1.z); ok[7]=(bf16)(accK[t][7]+bk1.w);
    *(v8bf*)(Qb + ((long)bidx * NPIX + n) * HID + d0) = oq;
    *(v8bf*)(Kb + ((long)bidx * NPIX + n) * HID + d0) = ok;
  }
}

// ---------------- pass A: row max m and row sum-exp l over all 4096 keys ----------------
__global__ void __launch_bounds__(256) k_stats(
    const bf16* __restrict__ Qb, const bf16* __restrict__ Kb,
    float* __restrict__ mArr, float* __restrict__ lArr)
{
  const int bidx = blockIdx.y;
  const int n0w  = blockIdx.x * 128 + (threadIdx.x >> 5) * 16;
  const int lane = threadIdx.x & 31;
  const int half = lane >> 4;
  const int l15  = lane & 15;

  const bf16* qrow = Qb + ((long)bidx * NPIX + n0w + l15) * HID;
  v16bf Aq0 = frag_cat(qrow + half * 8,      qrow + 16 + half * 8);
  v16bf Aq1 = frag_cat(qrow + 32 + half * 8, qrow + 48 + half * 8);

  float m_run[8], l_run[8];
  #pragma unroll
  for (int r = 0; r < 8; ++r) { m_run[r] = -3.0e38f; l_run[r] = 0.f; }

  for (int k0 = 0; k0 < NPIX; k0 += 64) {
    v8f s[4];
    #pragma unroll
    for (int kt = 0; kt < 4; ++kt) s[kt] = vzero8();
    #pragma unroll
    for (int kt = 0; kt < 4; ++kt) {
      const bf16* krow = Kb + ((long)bidx * NPIX + k0 + kt * 16 + l15) * HID;
      v16bf B0 = frag_cat(krow + half * 8,      krow + 16 + half * 8);
      v16bf B1 = frag_cat(krow + 32 + half * 8, krow + 48 + half * 8);
      s[kt] = wmma_bf16(Aq0, B0, s[kt]);
      s[kt] = wmma_bf16(Aq1, B1, s[kt]);
    }
    #pragma unroll
    for (int r = 0; r < 8; ++r) {
      float bm = fmaxf(fmaxf(s[0][r], s[1][r]), fmaxf(s[2][r], s[3][r]));
      bm = red16_max(bm);
      float mn = fmaxf(m_run[r], bm);
      float e = __expf(s[0][r]-mn) + __expf(s[1][r]-mn) +
                __expf(s[2][r]-mn) + __expf(s[3][r]-mn);
      e = red16_sum(e);
      l_run[r] = l_run[r] * __expf(m_run[r] - mn) + e;
      m_run[r] = mn;
    }
  }
  #pragma unroll
  for (int r = 0; r < 8; ++r) {
    if (l15 == r) {
      long row = (long)bidx * NPIX + n0w + half * 8 + r;
      mArr[row] = m_run[r];
      lArr[row] = l_run[r];
    }
  }
}

// ---------------- pass B: O = softmax(S) * V^T, S recomputed ----------------
// WG: 128 queries (8 waves x 16) x 64 channels; K/V staged in LDS, double-buffered async.
__global__ void __launch_bounds__(256) k_apply(
    const bf16* __restrict__ Qb, const bf16* __restrict__ Kb, const bf16* __restrict__ Vb,
    const float* __restrict__ mArr, const float* __restrict__ lArr,
    float* __restrict__ Out)
{
  const int bidx  = blockIdx.z;
  const int cbase = blockIdx.y * 64;
  const int tid   = threadIdx.x;
  const int wave  = tid >> 5;
  const int n0w   = blockIdx.x * 128 + wave * 16;
  const int lane  = tid & 31;
  const int half  = lane >> 4;
  const int l15   = lane & 15;

  __shared__ bf16 ldsK[2][64 * 72];    // [key][d], padded stride 72
  __shared__ bf16 ldsV[2][64 * 72];    // [c][p],  padded stride 72
  __shared__ bf16 plds[8][16 * 72];    // per-wave P strip (16q x 64k)

  const long krowbase = (long)bidx * NPIX;        // + key  -> row of Kb
  const long vrowbase = (long)bidx * CH + cbase;  // + c    -> row of Vb

  const bf16* qrow = Qb + ((long)bidx * NPIX + n0w + l15) * HID;
  v16bf Aq0 = frag_cat(qrow + half * 8,      qrow + 16 + half * 8);
  v16bf Aq1 = frag_cat(qrow + 32 + half * 8, qrow + 48 + half * 8);

  float mrow[8], invl[8];
  #pragma unroll
  for (int r = 0; r < 8; ++r) {
    long row = (long)bidx * NPIX + n0w + half * 8 + r;
    mrow[r] = mArr[row];
    invl[r] = 1.0f / lArr[row];
  }

  v8f accO[4];
  #pragma unroll
  for (int i = 0; i < 4; ++i) accO[i] = vzero8();

  // stage one 64-key block of K (64x64) and V (64x64): 4 x 16B chunks per thread
  auto stage_kv = [&](bf16* lk, bf16* lv, int k0) {
    #pragma unroll
    for (int i = 0; i < 2; ++i) {
      int ch = tid * 2 + i;            // 0..511
      int key = ch >> 3, c8 = ch & 7;
      async_copy16(Kb + (krowbase + k0 + key) * HID + c8 * 8,
                   lk + key * 72 + c8 * 8);
    }
    #pragma unroll
    for (int i = 0; i < 2; ++i) {
      int ch = tid * 2 + i;            // 0..511
      int c = ch >> 3, p8 = ch & 7;
      async_copy16(Vb + (vrowbase + c) * NPIX + k0 + p8 * 8,
                   lv + c * 72 + p8 * 8);
    }
  };

  stage_kv(ldsK[0], ldsV[0], 0);

  for (int k0 = 0; k0 < NPIX; k0 += 64) {
    const int cur = (k0 >> 6) & 1;
    __syncthreads();                          // all reads of buffer cur^1 finished
    if (k0 + 64 < NPIX) {
      stage_kv(ldsK[cur ^ 1], ldsV[cur ^ 1], k0 + 64);
      WAIT_ASYNC(4);                          // current block's 4 ops done (in-order)
    } else {
      WAIT_ASYNC(0);
    }
    __syncthreads();                          // staged buffer cur visible WG-wide

    // S strip 16q x 64k from LDS K
    v8f s[4];
    #pragma unroll
    for (int kt = 0; kt < 4; ++kt) s[kt] = vzero8();
    #pragma unroll
    for (int kt = 0; kt < 4; ++kt) {
      const bf16* krow = &ldsK[cur][(kt * 16 + l15) * 72];
      v16bf B0 = frag_cat(krow + half * 8,      krow + 16 + half * 8);
      v16bf B1 = frag_cat(krow + 32 + half * 8, krow + 48 + half * 8);
      s[kt] = wmma_bf16(Aq0, B0, s[kt]);
      s[kt] = wmma_bf16(Aq1, B1, s[kt]);
    }

    // P = exp(S - m) / l -> wave-private LDS strip (D-layout -> A-layout); in-order DS,
    // no WG barrier needed.
    #pragma unroll
    for (int kt = 0; kt < 4; ++kt) {
      #pragma unroll
      for (int r = 0; r < 8; ++r) {
        float p = __expf(s[kt][r] - mrow[r]) * invl[r];
        plds[wave][(half * 8 + r) * 72 + kt * 16 + l15] = (bf16)p;
      }
    }
    const bf16* prow = &plds[wave][l15 * 72];
    v16bf Ap0 = frag_cat(prow + half * 8,      prow + 16 + half * 8);
    v16bf Ap1 = frag_cat(prow + 32 + half * 8, prow + 48 + half * 8);

    // O accumulation over 4 channel tiles from LDS V
    #pragma unroll
    for (int ct = 0; ct < 4; ++ct) {
      const bf16* vrow = &ldsV[cur][(ct * 16 + l15) * 72];
      v16bf B0 = frag_cat(vrow + half * 8,      vrow + 16 + half * 8);
      v16bf B1 = frag_cat(vrow + 32 + half * 8, vrow + 48 + half * 8);
      accO[ct] = wmma_bf16(Ap0, B0, accO[ct]);
      accO[ct] = wmma_bf16(Ap1, B1, accO[ct]);
    }
  }

  // store: lane holds fixed channel, 8 consecutive query pixels -> two float4 stores
  #pragma unroll
  for (int ct = 0; ct < 4; ++ct) {
    long c = cbase + ct * 16 + l15;
    float* o = Out + ((long)bidx * CH + c) * NPIX + n0w + half * 8;
    float4 lo, hi;
    lo.x = accO[ct][0]; lo.y = accO[ct][1]; lo.z = accO[ct][2]; lo.w = accO[ct][3];
    hi.x = accO[ct][4]; hi.y = accO[ct][5]; hi.z = accO[ct][6]; hi.w = accO[ct][7];
    *(float4*)o = lo;
    *(float4*)(o + 4) = hi;
  }
}

// ---------------- launch ----------------
extern "C" void kernel_launch(void* const* d_in, const int* in_sizes, int n_in,
                              void* d_out, int out_size, void* d_ws, size_t ws_size,
                              hipStream_t stream)
{
  const float* Xq = (const float*)d_in[0];   // query_features (b,c,h,w)
  const float* Xk = (const float*)d_in[1];   // key_features   (b,c,h,w)
  const float* Wq = (const float*)d_in[2];   // (64, 512)
  const float* bq = (const float*)d_in[3];   // (64,)
  const float* Wk = (const float*)d_in[4];   // (64, 512)
  const float* bk = (const float*)d_in[5];   // (64,)
  float* Out = (float*)d_out;
  (void)in_sizes; (void)n_in; (void)out_size; (void)ws_size;

  char* ws = (char*)d_ws;
  bf16* Qb  = (bf16*)ws;  ws += (size_t)BATCH * NPIX * HID * 2;
  bf16* Kb  = (bf16*)ws;  ws += (size_t)BATCH * NPIX * HID * 2;
  bf16* Vb  = (bf16*)ws;  ws += (size_t)BATCH * CH * NPIX * 2;
  bf16* Wqb = (bf16*)ws;  ws += (size_t)HID * CH * 2;
  bf16* Wkb = (bf16*)ws;  ws += (size_t)HID * CH * 2;
  float* mArr = (float*)ws; ws += (size_t)BATCH * NPIX * 4;
  float* lArr = (float*)ws; ws += (size_t)BATCH * NPIX * 4;

  {
    long quads = (long)BATCH * CH * NPIX / 4;
    k_cast_v<<<dim3((unsigned)((quads + 255) / 256)), 256, 0, stream>>>(Xk, Vb);
  }
  k_cast_w<<<dim3((2 * HID * CH / 4 + 255) / 256), 256, 0, stream>>>(Wq, Wk, Wqb, Wkb);

  dim3 gproj(NPIX / 128, BATCH);
  k_proj<<<gproj, 256, 0, stream>>>(Xq, Xk, Wqb, Wkb, bq, bk, Qb, Kb);

  dim3 gstats(NPIX / 128, BATCH);
  k_stats<<<gstats, 256, 0, stream>>>(Qb, Kb, mArr, lArr);

  dim3 gapply(NPIX / 128, CH / 64, BATCH);
  k_apply<<<gapply, 256, 0, stream>>>(Qb, Kb, Vb, mArr, lArr, Out);
}
